// Model_41360535060511
// MI455X (gfx1250) — matrix operand order repaired and checked by reading, not compiled, not run on hardware
//
#include <hip/hip_runtime.h>
#include <hip/hip_bf16.h>
#include <math.h>

// ---------------- problem constants ----------------
#define BB   8
#define VV   12288
#define EE   147456
#define C_FM 384
#define C_U  61
#define C_IN 64
#define C_HID 128
#define C_OUT 32

typedef __attribute__((ext_vector_type(16))) _Float16 v16h;
typedef __attribute__((ext_vector_type(8)))  _Float16 v8h;
typedef __attribute__((ext_vector_type(8)))  float    v8f;

// ---------------- fragment loaders (CDNA5 16-bit WMMA layouts) ----------------
// A 16x32 (MxK): lane M = lane&15, half = lane>>4.
// af[0..7]  = row[kk*32 + half*8 .. +8)
// af[8..15] = row[kk*32 + 16 + half*8 .. +8)
__device__ __forceinline__ v16h load_a_frag(const _Float16* __restrict__ row,
                                            int kk, int half) {
    const v8h lo = *(const v8h*)(row + kk * 32 + half * 8);
    const v8h hi = *(const v8h*)(row + kk * 32 + 16 + half * 8);
    v16h r;
#pragma unroll
    for (int i = 0; i < 8; ++i) { r[i] = lo[i]; r[8 + i] = hi[i]; }
    return r;
}

// B 32x16 (KxN): lane N = lane&15, half = lane>>4.
// bf[0..7]  = col[kk*32 + half*8 .. +8), bf[8..15] = col[kk*32 + 16 + half*8 .. +8)  (col stored K-major)
__device__ __forceinline__ v16h load_b_frag(const _Float16* __restrict__ col,
                                            int kk, int half) {
    const v8h lo = *(const v8h*)(col + kk * 32 + half * 8);
    const v8h hi = *(const v8h*)(col + kk * 32 + 16 + half * 8);
    v16h r;
#pragma unroll
    for (int i = 0; i < 8; ++i) { r[i] = lo[i]; r[8 + i] = hi[i]; }
    return r;
}

// ---------------- helpers ----------------
__device__ __forceinline__ float bilin_zero(const float* __restrict__ img,
                                            int H, int W, float gx, float gy) {
    float xf = (gx + 1.f) * 0.5f * (float)W - 0.5f;
    float yf = (gy + 1.f) * 0.5f * (float)H - 0.5f;
    float x0f = floorf(xf), y0f = floorf(yf);
    float wx1 = xf - x0f, wx0 = 1.f - wx1;
    float wy1 = yf - y0f, wy0 = 1.f - wy1;
    int x0 = (int)x0f, y0 = (int)y0f, x1 = x0 + 1, y1 = y0 + 1;
    float acc = 0.f;
    if (y0 >= 0 && y0 < H) {
        if (x0 >= 0 && x0 < W) acc += wy0 * wx0 * img[y0 * W + x0];
        if (x1 >= 0 && x1 < W) acc += wy0 * wx1 * img[y0 * W + x1];
    }
    if (y1 >= 0 && y1 < H) {
        if (x0 >= 0 && x0 < W) acc += wy1 * wx0 * img[y1 * W + x0];
        if (x1 >= 0 && x1 < W) acc += wy1 * wx1 * img[y1 * W + x1];
    }
    return acc;
}

// ---------------- weight pre-conversion (once per launch) ----------------
// cw_h:  [64][384] f16 row-major, rows >= 61 zero  (A-matrix for conv GEMM)
// w1t_h: [128][64] f16  (W1 transposed: [n][k], B-matrix cols for GEMM1)
// w2t_h: [32][128] f16  (W2 transposed: [n][k], B-matrix cols for GEMM2)
__global__ void wcvt_kernel(const float* __restrict__ conv_w,
                            const float* __restrict__ W1,
                            const float* __restrict__ W2,
                            _Float16* __restrict__ cw_h,
                            _Float16* __restrict__ w1t_h,
                            _Float16* __restrict__ w2t_h) {
    const int i = blockIdx.x * blockDim.x + threadIdx.x;
    if (i < 64 * C_FM) {
        const int r = i / C_FM, c = i % C_FM;
        cw_h[i] = (r < C_U) ? (_Float16)conv_w[r * C_FM + c] : (_Float16)0.f;
    }
    if (i < C_HID * C_IN) {
        const int n = i / C_IN, k = i % C_IN;
        w1t_h[i] = (_Float16)W1[k * C_HID + n];
    }
    if (i < C_OUT * C_HID) {
        const int n = i / C_HID, k = i % C_HID;
        w2t_h[i] = (_Float16)W2[k * C_OUT + n];
    }
}

// ---------------- degree count ----------------
__global__ void deg_kernel(const int* __restrict__ erow, float* __restrict__ deg) {
    int e = blockIdx.x * blockDim.x + threadIdx.x;
    if (e < EE) atomicAdd(&deg[erow[e]], 1.f);
}

// ---------------- fused sampler + conv GEMM ----------------
// grid: BB * (VV/16) blocks, 128 threads (4 waves; wave = M-tile of padded conv_w).
__global__ void vertex_feat_kernel(const float* __restrict__ fm,
                                   const float* __restrict__ dm,
                                   const float* __restrict__ verts,
                                   const float* __restrict__ fz,
                                   const float* __restrict__ cz,
                                   const _Float16* __restrict__ cw_h,
                                   const float* __restrict__ conv_b,
                                   float* __restrict__ x_buf) {
    const int blk = blockIdx.x;
    const int b  = blk / (VV / 16);
    const int v0 = (blk % (VV / 16)) * 16;

    __shared__ float    s_w[16][4];
    __shared__ int      s_i[16][4];
    __shared__ __align__(16) _Float16 s_sf[16][C_FM];  // [vertex n][channel k]

    const int t = threadIdx.x;

    if (t < 16) {
        const int v = v0 + t;
        const float* vp = verts + ((size_t)b * VV + v) * 3;
        const float fzb = fz[b];
        const float czb = cz[b * 2 + 0];
        const float px = fzb * vp[0] + czb;
        const float py = fzb * vp[1] + czb;
        const float pz = fzb * (vp[2] + 100.f);

        const float d = bilin_zero(dm + (size_t)b * 128 * 128, 128, 128, px, py);
        const float vis = (fabsf(d - pz) < 0.005f) ? 1.f : 0.f;

        const float xf = (px + 1.f) * 32.f - 0.5f;
        const float yf = (py + 1.f) * 32.f - 0.5f;
        const float x0f = floorf(xf), y0f = floorf(yf);
        const float wx1 = xf - x0f, wx0 = 1.f - wx1;
        const float wy1 = yf - y0f, wy0 = 1.f - wy1;
        const int x0 = (int)x0f, y0 = (int)y0f, x1 = x0 + 1, y1 = y0 + 1;
        const bool bx0 = (x0 >= 0) & (x0 < 64), bx1 = (x1 >= 0) & (x1 < 64);
        const bool by0 = (y0 >= 0) & (y0 < 64), by1 = (y1 >= 0) & (y1 < 64);
        s_w[t][0] = wy0 * wx0 * ((by0 & bx0) ? vis : 0.f);
        s_w[t][1] = wy0 * wx1 * ((by0 & bx1) ? vis : 0.f);
        s_w[t][2] = wy1 * wx0 * ((by1 & bx0) ? vis : 0.f);
        s_w[t][3] = wy1 * wx1 * ((by1 & bx1) ? vis : 0.f);
        s_i[t][0] = min(max(y0, 0), 63);
        s_i[t][1] = min(max(y1, 0), 63);
        s_i[t][2] = min(max(x0, 0), 63);
        s_i[t][3] = min(max(x1, 0), 63);
    }
    __syncthreads();

    // gather sf tile (384 channels x 16 vertices) into LDS as f16
    {
        const int n  = t & 15;
        const int cg = t >> 4;
        const float w0 = s_w[n][0], w1 = s_w[n][1], w2 = s_w[n][2], w3 = s_w[n][3];
        const int iy0 = s_i[n][0], iy1 = s_i[n][1], ix0 = s_i[n][2], ix1 = s_i[n][3];
        const float* fmb = fm + (size_t)b * C_FM * 64 * 64;
#pragma unroll 4
        for (int c = cg; c < C_FM; c += 8) {
            const float* p = fmb + (size_t)c * 64 * 64;
            float val = w0 * p[iy0 * 64 + ix0] + w1 * p[iy0 * 64 + ix1] +
                        w2 * p[iy1 * 64 + ix0] + w3 * p[iy1 * 64 + ix1];
            s_sf[n][c] = (_Float16)val;
        }
    }
    __syncthreads();

    // WMMA: D(64x16) = cw_h(64x384) * sf(384x16)
    const int wave = t >> 5;
    const int lane = t & 31;
    const int nl   = lane & 15;
    const int half = lane >> 4;
    const _Float16* arow = cw_h + (size_t)(wave * 16 + nl) * C_FM;
    const _Float16* bcol = &s_sf[nl][0];
    v8f acc = {};
#pragma unroll
    for (int kk = 0; kk < C_FM / 32; ++kk) {
        const v16h af = load_a_frag(arow, kk, half);
        const v16h bf = load_b_frag(bcol, kk, half);
        acc = __builtin_amdgcn_wmma_f32_16x16x32_f16(false, af, false, bf,
                                                     (short)0, acc, false, false);
    }
    // D: lane col N = lane&15 (vertex), VGPR r -> row M = half*8 + r
    {
        const size_t rowbase = ((size_t)b * VV + v0 + nl) * C_IN;
#pragma unroll
        for (int r = 0; r < 8; ++r) {
            const int M = wave * 16 + half * 8 + r;
            if (M < C_U) x_buf[rowbase + M] = acc[r] + conv_b[M];
        }
    }
    if (t < 16) {
        const float* vp = verts + ((size_t)b * VV + v0 + t) * 3;
        const size_t rb = ((size_t)b * VV + v0 + t) * C_IN;
        x_buf[rb + 61] = vp[0];
        x_buf[rb + 62] = vp[1];
        x_buf[rb + 63] = vp[2];
    }
}

// ---------------- edge aggregation (segment sum via atomics) ----------------
template <int F>
__global__ void agg_kernel(const int* __restrict__ erow, const int* __restrict__ ecol,
                           const float* __restrict__ src, float* __restrict__ dst) {
    const int wid   = (blockIdx.x * blockDim.x + threadIdx.x) >> 5;
    const int lane  = threadIdx.x & 31;
    const int nwave = (gridDim.x * blockDim.x) >> 5;
    for (int e = wid; e < EE; e += nwave) {
        const int r = erow[e];
        const int c = ecol[e];
        __builtin_prefetch(src + (size_t)c * F, 0, 1);   // global_prefetch_b8
        for (int b = 0; b < BB; ++b) {
            const float* s = src + ((size_t)b * VV + c) * F;
            float*       d = dst + ((size_t)b * VV + r) * F;
#pragma unroll
            for (int j = 0; j < F / 32; ++j)
                atomicAdd(&d[lane + 32 * j], s[lane + 32 * j]);
        }
    }
}

// ---------------- GEMM1: h = relu((xagg/deg) @ W1 + b1) ----------------
// grid: (B*V)/16 blocks, 256 threads (8 waves = 8 N-tiles of 16).
__global__ void gemm1_kernel(const float* __restrict__ xagg, const float* __restrict__ deg,
                             const _Float16* __restrict__ w1t_h, const float* __restrict__ b1,
                             float* __restrict__ h) {
    __shared__ __align__(16) _Float16 s_a[16][C_IN];
    const int t = threadIdx.x;
    const size_t r0 = (size_t)blockIdx.x * 16;
    for (int i = t; i < 16 * C_IN; i += 256) {
        const int mr = i >> 6, k = i & 63;
        const size_t row = r0 + mr;
        const float dg = fmaxf(deg[(int)(row % VV)], 1.f);
        s_a[mr][k] = (_Float16)(xagg[row * C_IN + k] / dg);
    }
    __syncthreads();
    const int wave = t >> 5, lane = t & 31;
    const int nl = lane & 15, half = lane >> 4;
    const int ncol = wave * 16 + nl;
    const _Float16* arow = &s_a[nl][0];
    const _Float16* bcol = w1t_h + (size_t)ncol * C_IN;
    v8f acc = {};
#pragma unroll
    for (int kk = 0; kk < C_IN / 32; ++kk) {
        const v16h af = load_a_frag(arow, kk, half);
        const v16h bf = load_b_frag(bcol, kk, half);
        acc = __builtin_amdgcn_wmma_f32_16x16x32_f16(false, af, false, bf,
                                                     (short)0, acc, false, false);
    }
    const float bias = b1[ncol];
#pragma unroll
    for (int r = 0; r < 8; ++r) {
        const int mr = half * 8 + r;
        const float v = acc[r] + bias;
        h[(r0 + mr) * C_HID + ncol] = v > 0.f ? v : 0.f;
    }
}

// ---------------- GEMM2 + dv + output assembly ----------------
// grid: (B*V)/16 blocks, 64 threads (2 waves = 2 N-tiles of W2).
__global__ void gemm2_kernel(const float* __restrict__ hagg, const float* __restrict__ h,
                             const float* __restrict__ deg, const _Float16* __restrict__ w2t_h,
                             const float* __restrict__ W_dv, const float* __restrict__ verts,
                             float* __restrict__ out) {
    __shared__ __align__(16) _Float16 s_a[16][C_HID];
    const int t = threadIdx.x;
    const size_t r0 = (size_t)blockIdx.x * 16;
    for (int i = t; i < 16 * C_HID; i += 64) {
        const int mr = i >> 7, k = i & 127;
        const size_t row = r0 + mr;
        const float dg = fmaxf(deg[(int)(row % VV)], 1.f);
        s_a[mr][k] = (_Float16)(hagg[row * C_HID + k] / dg);
    }
    __syncthreads();
    const int wave = t >> 5, lane = t & 31;
    const int nl = lane & 15, half = lane >> 4;
    const int ncol = wave * 16 + nl;
    const _Float16* arow = &s_a[nl][0];
    const _Float16* bcol = w2t_h + (size_t)ncol * C_HID;
    v8f acc = {};
#pragma unroll
    for (int kk = 0; kk < C_HID / 32; ++kk) {
        const v16h af = load_a_frag(arow, kk, half);
        const v16h bf = load_b_frag(bcol, kk, half);
        acc = __builtin_amdgcn_wmma_f32_16x16x32_f16(false, af, false, bf,
                                                     (short)0, acc, false, false);
    }
#pragma unroll
    for (int r = 0; r < 8; ++r) {
        const int mr = half * 8 + r;
        out[(r0 + mr) * 38 + ncol] = acc[r];
    }
    // dv = h @ W_dv (3 cols), vertex outputs (threads 0..47: n=t%16, i=t/16)
    if (t < 48) {
        const int n = t & 15, i = t >> 4;
        const size_t row = r0 + n;
        const float* hr = h + row * C_HID;
        float s = 0.f;
        for (int k = 0; k < C_HID; ++k) s += hr[k] * W_dv[k * 3 + i];
        const float vi = verts[row * 3 + i];
        out[row * 38 + 32 + i] = (vi + s) * 1000.f;
        out[row * 38 + 35 + i] = s * 1000.f;
    }
}

// ---------------- launch ----------------
extern "C" void kernel_launch(void* const* d_in, const int* in_sizes, int n_in,
                              void* d_out, int out_size, void* d_ws, size_t ws_size,
                              hipStream_t stream) {
    const float* fm     = (const float*)d_in[0];
    const float* dm     = (const float*)d_in[1];
    const float* verts  = (const float*)d_in[2];
    const float* fz     = (const float*)d_in[3];
    const float* cz     = (const float*)d_in[4];
    const float* conv_w = (const float*)d_in[5];
    const float* conv_b = (const float*)d_in[6];
    const float* W1     = (const float*)d_in[7];
    const float* b1     = (const float*)d_in[8];
    const float* W2     = (const float*)d_in[9];
    const float* W_dv   = (const float*)d_in[10];
    const int*   erow   = (const int*)d_in[11];
    const int*   ecol   = (const int*)d_in[12];
    float* out = (float*)d_out;

    const size_t SX = (size_t)BB * VV * C_IN  * sizeof(float);   // 24 MB
    const size_t SH = (size_t)BB * VV * C_HID * sizeof(float);   // 48 MB
    const size_t SD = (size_t)VV * sizeof(float);
    char* ws = (char*)d_ws;
    float*    x     = (float*)(ws);
    float*    xagg  = (float*)(ws + SX);
    float*    h     = (float*)(ws + 2 * SX);
    float*    hagg  = (float*)(ws + 2 * SX + SH);
    float*    deg   = (float*)(ws + 2 * SX + 2 * SH);
    _Float16* cw_h  = (_Float16*)(ws + 2 * SX + 2 * SH + SD);
    _Float16* w1t_h = cw_h + 64 * C_FM;
    _Float16* w2t_h = w1t_h + C_HID * C_IN;

    (void)hipMemsetAsync(xagg, 0, SX, stream);
    (void)hipMemsetAsync(hagg, 0, SH, stream);
    (void)hipMemsetAsync(deg,  0, SD, stream);

    wcvt_kernel<<<(64 * C_FM + 255) / 256, 256, 0, stream>>>(conv_w, W1, W2,
                                                             cw_h, w1t_h, w2t_h);
    deg_kernel<<<(EE + 255) / 256, 256, 0, stream>>>(erow, deg);

    vertex_feat_kernel<<<BB * (VV / 16), 128, 0, stream>>>(
        fm, dm, verts, fz, cz, cw_h, conv_b, x);

    agg_kernel<C_IN><<<2304, 256, 0, stream>>>(erow, ecol, x, xagg);

    gemm1_kernel<<<(BB * VV) / 16, 256, 0, stream>>>(xagg, deg, w1t_h, b1, h);

    agg_kernel<C_HID><<<2304, 256, 0, stream>>>(erow, ecol, h, hagg);

    gemm2_kernel<<<(BB * VV) / 16, 64, 0, stream>>>(hagg, h, deg, w2t_h, W_dv, verts, out);
}
